// Reconstruction_7052336300053
// MI455X (gfx1250) — compile-verified, matrix operand order checked by reading, not run on hardware
//
#include <hip/hip_runtime.h>

typedef __attribute__((ext_vector_type(16))) _Float16 v16h;
typedef __attribute__((ext_vector_type(8)))  float    v8f;
typedef __attribute__((ext_vector_type(8)))  _Float16 v8h;
typedef __attribute__((ext_vector_type(2)))  _Float16 v2h;

// sigmoid(d - 0.3) = 1 / (1 + exp(0.3 - d))
__device__ __forceinline__ float sigm(float d) {
    return 1.0f / (1.0f + __expf(0.3f - d));
}

// ---------------------------------------------------------------------------
// Pass 1: bake a z-pair table. tab[v] (u32) = packed( f16 sig(den[v]),
// f16 sig(den[v+1]) ), with the z==255 "next" padded to 0. One aligned dword
// gather then serves BOTH z-corners of a trilinear sample. 64 MiB: resident
// in the 192 MB L2, so steady-state gathers never touch HBM.
// ---------------------------------------------------------------------------
__global__ __launch_bounds__(256) void sig_pack_kernel(const float* __restrict__ den,
                                                       unsigned* __restrict__ tab) {
    size_t g = (size_t)blockIdx.x * 256 + threadIdx.x;   // group of 4 voxels
    const float4* d4 = (const float4*)den;
    float4 v = d4[g];
    int z3 = (int)((4 * g + 3) & 255);                   // z of 4th voxel
    size_t nIdx = (z3 != 255) ? (4 * g + 4) : (4 * g + 3); // safe in-bounds
    float nxt = den[nIdx];
    float s0 = sigm(v.x), s1 = sigm(v.y), s2 = sigm(v.z), s3 = sigm(v.w);
    float s4 = (z3 != 255) ? sigm(nxt) : 0.0f;
    v8h o;
    o[0] = (_Float16)s0; o[1] = (_Float16)s1;
    o[2] = (_Float16)s1; o[3] = (_Float16)s2;
    o[4] = (_Float16)s2; o[5] = (_Float16)s3;
    o[6] = (_Float16)s3; o[7] = (_Float16)s4;
    ((v8h*)tab)[g] = o;
}

// ---------------------------------------------------------------------------
// Branchless trilinear from the z-pair table: 4 unconditional aligned dword
// gathers (clamped addresses), invalid corners zeroed by cndmask. Matches
// map_coordinates(order=1, mode="constant", cval=0).
// ---------------------------------------------------------------------------
__device__ __forceinline__ float trilin_tab(const unsigned* __restrict__ tab,
                                            float px, float py, float pz) {
    float fx = floorf(px), fy = floorf(py), fz = floorf(pz);
    int ix = (int)fx, iy = (int)fy, iz = (int)fz;
    float wx = px - fx, wy = py - fy, wz = pz - fz;
    bool bx0 = (unsigned)ix        < 256u, bx1 = (unsigned)(ix + 1) < 256u;
    bool by0 = (unsigned)iy        < 256u, by1 = (unsigned)(iy + 1) < 256u;
    bool bz0 = (unsigned)iz        < 256u, bz1 = (unsigned)(iz + 1) < 256u;
    int xc0 = min(max(ix,     0), 255), xc1 = min(max(ix + 1, 0), 255);
    int yc0 = min(max(iy,     0), 255), yc1 = min(max(iy + 1, 0), 255);
    int zc  = min(max(iz,     0), 255);
    unsigned q00 = tab[(((xc0 << 8) | yc0) << 8) | zc];   // (x0,y0): v[z], v[z+1]
    unsigned q01 = tab[(((xc0 << 8) | yc1) << 8) | zc];
    unsigned q10 = tab[(((xc1 << 8) | yc0) << 8) | zc];
    unsigned q11 = tab[(((xc1 << 8) | yc1) << 8) | zc];
    bool zlo = (iz < 0);                                  // iz==-1 -> v[z+1]=v[0]=lo half
    v2h h00 = __builtin_bit_cast(v2h, q00);
    v2h h01 = __builtin_bit_cast(v2h, q01);
    v2h h10 = __builtin_bit_cast(v2h, q10);
    v2h h11 = __builtin_bit_cast(v2h, q11);
    float a00 = (float)h00[0], c00n = zlo ? (float)h00[0] : (float)h00[1];
    float a01 = (float)h01[0], c01n = zlo ? (float)h01[0] : (float)h01[1];
    float a10 = (float)h10[0], c10n = zlo ? (float)h10[0] : (float)h10[1];
    float a11 = (float)h11[0], c11n = zlo ? (float)h11[0] : (float)h11[1];
    float v000 = (bx0 && by0 && bz0) ? a00  : 0.0f;
    float v001 = (bx0 && by0 && bz1) ? c00n : 0.0f;
    float v010 = (bx0 && by1 && bz0) ? a01  : 0.0f;
    float v011 = (bx0 && by1 && bz1) ? c01n : 0.0f;
    float v100 = (bx1 && by0 && bz0) ? a10  : 0.0f;
    float v101 = (bx1 && by0 && bz1) ? c10n : 0.0f;
    float v110 = (bx1 && by1 && bz0) ? a11  : 0.0f;
    float v111 = (bx1 && by1 && bz1) ? c11n : 0.0f;
    float c00 = v000 + wx * (v100 - v000);
    float c01 = v001 + wx * (v101 - v001);
    float c10 = v010 + wx * (v110 - v010);
    float c11 = v011 + wx * (v111 - v011);
    float c0  = c00  + wy * (c10 - c00);
    float c1  = c01  + wy * (c11 - c01);
    return c0 + wz * (c1 - c0);
}

// Fallback (workspace too small): branchless, 8 clamped f32 loads + inline
// sigmoid per corner.
__device__ __forceinline__ float trilin_raw(const float* __restrict__ den,
                                            float px, float py, float pz) {
    float fx = floorf(px), fy = floorf(py), fz = floorf(pz);
    int ix = (int)fx, iy = (int)fy, iz = (int)fz;
    float wx = px - fx, wy = py - fy, wz = pz - fz;
    bool bx0 = (unsigned)ix        < 256u, bx1 = (unsigned)(ix + 1) < 256u;
    bool by0 = (unsigned)iy        < 256u, by1 = (unsigned)(iy + 1) < 256u;
    bool bz0 = (unsigned)iz        < 256u, bz1 = (unsigned)(iz + 1) < 256u;
    int xc0 = min(max(ix,     0), 255), xc1 = min(max(ix + 1, 0), 255);
    int yc0 = min(max(iy,     0), 255), yc1 = min(max(iy + 1, 0), 255);
    int zc0 = min(max(iz,     0), 255), zc1 = min(max(iz + 1, 0), 255);
    int r00 = ((xc0 << 8) | yc0) << 8, r01 = ((xc0 << 8) | yc1) << 8;
    int r10 = ((xc1 << 8) | yc0) << 8, r11 = ((xc1 << 8) | yc1) << 8;
    float d000 = den[r00 | zc0], d001 = den[r00 | zc1];
    float d010 = den[r01 | zc0], d011 = den[r01 | zc1];
    float d100 = den[r10 | zc0], d101 = den[r10 | zc1];
    float d110 = den[r11 | zc0], d111 = den[r11 | zc1];
    float v000 = (bx0 && by0 && bz0) ? sigm(d000) : 0.0f;
    float v001 = (bx0 && by0 && bz1) ? sigm(d001) : 0.0f;
    float v010 = (bx0 && by1 && bz0) ? sigm(d010) : 0.0f;
    float v011 = (bx0 && by1 && bz1) ? sigm(d011) : 0.0f;
    float v100 = (bx1 && by0 && bz0) ? sigm(d100) : 0.0f;
    float v101 = (bx1 && by0 && bz1) ? sigm(d101) : 0.0f;
    float v110 = (bx1 && by1 && bz0) ? sigm(d110) : 0.0f;
    float v111 = (bx1 && by1 && bz1) ? sigm(d111) : 0.0f;
    float c00 = v000 + wx * (v100 - v000);
    float c01 = v001 + wx * (v101 - v001);
    float c10 = v010 + wx * (v110 - v010);
    float c11 = v011 + wx * (v111 - v011);
    float c0  = c00  + wy * (c10 - c00);
    float c1  = c01  + wy * (c11 - c01);
    return c0 + wz * (c1 - c0);
}

// ---------------------------------------------------------------------------
// Pass 2: one wave32 integrates 16 rays. 16x32 f16 A-tile of samples per
// chunk; B = ones(32x16); v_wmma_f32_16x16x32_f16 accumulates the per-ray
// sample sums in f32. Chunk bounds are readfirstlane'd into SGPRs so the
// loop is a scalar loop: zero EXEC churn, and EXEC==all-1s at the WMMA by
// construction (ISA 7.12 requirement).
// ---------------------------------------------------------------------------
template <bool TAB>
__global__ __launch_bounds__(128) void drr_kernel(const float* __restrict__ den,
                                                  const unsigned* __restrict__ tab,
                                                  const float* __restrict__ src,
                                                  const float* __restrict__ tgt,
                                                  const float* __restrict__ aff,
                                                  const int* __restrict__ npts,
                                                  float* __restrict__ out,
                                                  int N) {
    int tid  = threadIdx.x;
    int wave = tid >> 5;
    int lane = tid & 31;
    int gwave = blockIdx.x * (blockDim.x >> 5) + wave;
    int rayBase = gwave << 4;            // 16 rays per wave
    if (rayBase >= N) return;            // whole-wave exit

    int m  = lane & 15;
    int hi = lane >> 4;
    int ray = rayBase + m;
    if (ray >= N) ray = N - 1;           // clamp; store guarded below

    int P = npts[0];
    float invPm1 = 1.0f / (float)(P - 1);

    // Inverse of 4x4 affine [M|t]: inv3 = adj(M)/det, b = -inv3*t
    float m00 = aff[0], m01 = aff[1], m02 = aff[2],  t0 = aff[3];
    float m10 = aff[4], m11 = aff[5], m12 = aff[6],  t1 = aff[7];
    float m20 = aff[8], m21 = aff[9], m22 = aff[10], t2 = aff[11];
    float det = m00 * (m11 * m22 - m12 * m21)
              - m01 * (m10 * m22 - m12 * m20)
              + m02 * (m10 * m21 - m11 * m20);
    float rd = 1.0f / det;
    float i00 = (m11*m22 - m12*m21) * rd, i01 = (m02*m21 - m01*m22) * rd, i02 = (m01*m12 - m02*m11) * rd;
    float i10 = (m12*m20 - m10*m22) * rd, i11 = (m00*m22 - m02*m20) * rd, i12 = (m02*m10 - m00*m12) * rd;
    float i20 = (m10*m21 - m11*m20) * rd, i21 = (m01*m20 - m00*m21) * rd, i22 = (m00*m11 - m01*m10) * rd;
    float bx = -(i00 * t0 + i01 * t1 + i02 * t2);
    float by = -(i10 * t0 + i11 * t1 + i12 * t2);
    float bz = -(i20 * t0 + i21 * t1 + i22 * t2);

    // Ray endpoints (world) -> length -> voxel-index space
    const float* s3 = src + 3 * ray;
    const float* t3 = tgt + 3 * ray;
    float sxw = s3[0], syw = s3[1], szw = s3[2];
    float txw = t3[0], tyw = t3[1], tzw = t3[2];
    float dxw = txw - sxw, dyw = tyw - syw, dzw = tzw - szw;
    float raylen = sqrtf(dxw*dxw + dyw*dyw + dzw*dzw);

    float svx = i00*sxw + i01*syw + i02*szw + bx;
    float svy = i10*sxw + i11*syw + i12*szw + by;
    float svz = i20*sxw + i21*syw + i22*szw + bz;
    float tvx = i00*txw + i01*tyw + i02*tzw + bx;
    float tvy = i10*txw + i11*tyw + i12*tzw + by;
    float tvz = i20*txw + i21*tyw + i22*tzw + bz;
    float dvx = tvx - svx, dvy = tvy - svy, dvz = tvz - svz;

    // Per-ray slab clip against the padded box (-1,256)^3 in alpha space
    float amin = 0.0f, amax = 1.0f;
    {
        float sv[3] = {svx, svy, svz};
        float dv[3] = {dvx, dvy, dvz};
        #pragma unroll
        for (int a = 0; a < 3; ++a) {
            if (fabsf(dv[a]) > 1e-12f) {
                float r  = 1.0f / dv[a];
                float ta = (-1.0f  - sv[a]) * r;
                float tb = (256.0f - sv[a]) * r;
                amin = fmaxf(amin, fminf(ta, tb));
                amax = fminf(amax, fmaxf(ta, tb));
            } else if (sv[a] <= -1.0f || sv[a] >= 256.0f) {
                amin = 1e30f; amax = -1e30f;    // ray misses the box
            }
        }
    }
    // Wave-wide union of sample ranges -> uniform chunk bounds
    #pragma unroll
    for (int off = 16; off > 0; off >>= 1) {
        amin = fminf(amin, __shfl_xor(amin, off, 32));
        amax = fmaxf(amax, __shfl_xor(amax, off, 32));
    }
    // Branchless bounds: a miss (amin>amax) saturates to cB huge / cE negative
    // -> zero iterations. Scalarize so the loop has SGPR control flow.
    int iFirst = (int)floorf(fmaxf(amin, 0.0f) * (float)(P - 1));
    int iLast  = (int)ceilf (fminf(amax, 1.0f) * (float)(P - 1));
    int cB = (max(iFirst, 0) >> 5);
    int cE = (min(iLast, P - 1) >> 5);
    cB = __builtin_amdgcn_readfirstlane(cB);
    cE = __builtin_amdgcn_readfirstlane(cE);

    v16h ones;
    #pragma unroll
    for (int h = 0; h < 16; ++h) ones[h] = (_Float16)1.0f;

    v8f acc = {};
    for (int c = cB; c <= cE; ++c) {
        v16h a;
        #pragma unroll
        for (int j = 0; j < 8; ++j) {
            int kb = ((j < 4) ? (2 * j) : (16 + 2 * (j - 4))) + (hi << 3);
            int i0 = (c << 5) + kb;
            float al0 = (float)i0       * invPm1;
            float al1 = (float)(i0 + 1) * invPm1;
            float s0, s1;
            if (TAB) {
                s0 = trilin_tab(tab, svx + al0*dvx, svy + al0*dvy, svz + al0*dvz);
                s1 = trilin_tab(tab, svx + al1*dvx, svy + al1*dvy, svz + al1*dvz);
            } else {
                s0 = trilin_raw(den, svx + al0*dvx, svy + al0*dvy, svz + al0*dvz);
                s1 = trilin_raw(den, svx + al1*dvx, svy + al1*dvy, svz + al1*dvz);
            }
            s0 = (i0     < P) ? s0 : 0.0f;                // branchless tail mask
            s1 = (i0 + 1 < P) ? s1 : 0.0f;
            a[2 * j]     = (_Float16)s0;
            a[2 * j + 1] = (_Float16)s1;
        }
        acc = __builtin_amdgcn_wmma_f32_16x16x32_f16(
            false, a, false, ones, (short)0, acc, false, false);
    }

    // Row M of D lives in VGPR (M%8), half-wave (M/8). Scale by raylen/P.
    float scaleMe = raylen / (float)P;    // lane m holds ray (rayBase+m)'s scale
    float res[8];
    #pragma unroll
    for (int v = 0; v < 8; ++v) {
        int rv = v + (hi << 3);
        float sc = __shfl(scaleMe, rv, 32);
        res[v] = acc[v] * sc;
    }
    if ((lane & 15) == 0) {               // lanes 0 (rays 0-7) and 16 (rays 8-15)
        #pragma unroll
        for (int v = 0; v < 8; ++v) {
            int r = rayBase + v + (hi << 3);
            if (r < N) out[r] = res[v];
        }
    }
}

extern "C" void kernel_launch(void* const* d_in, const int* in_sizes, int n_in,
                              void* d_out, int out_size, void* d_ws, size_t ws_size,
                              hipStream_t stream) {
    const float* den = (const float*)d_in[0];   // [256,256,256] f32
    const float* src = (const float*)d_in[1];   // [1,N,3] f32
    const float* tgt = (const float*)d_in[2];   // [1,N,3] f32
    const float* aff = (const float*)d_in[3];   // [4,4] f32
    const int*   np  = (const int*)d_in[4];     // scalar n_points
    float* out = (float*)d_out;

    int N = in_sizes[1] / 3;
    int waves  = (N + 15) / 16;
    int blocks = (waves + 3) / 4;               // 4 waves (128 threads) per block

    const size_t tabBytes = (size_t)256 * 256 * 256 * sizeof(unsigned); // 64 MiB
    if (ws_size >= tabBytes) {
        unsigned* tab = (unsigned*)d_ws;
        sig_pack_kernel<<<16384, 256, 0, stream>>>(den, tab);
        drr_kernel<true><<<blocks, 128, 0, stream>>>(den, tab, src, tgt, aff, np, out, N);
    } else {
        drr_kernel<false><<<blocks, 128, 0, stream>>>(den, nullptr, src, tgt, aff, np, out, N);
    }
}
